// SWAWrapper_32031866094018
// MI455X (gfx1250) — matrix operand order repaired and checked by reading, not compiled, not run on hardware
//
#include <hip/hip_runtime.h>
#include <stdint.h>

#define H   16
#define HKV 4
#define HD  128

typedef __bf16 bf16;
typedef __attribute__((ext_vector_type(16))) __bf16 v16bf;
typedef __attribute__((ext_vector_type(8)))  __bf16 v8bf;
typedef __attribute__((ext_vector_type(8)))  float  v8f;
typedef unsigned int v4u  __attribute__((ext_vector_type(4)));
typedef int          v4i  __attribute__((ext_vector_type(4)));
typedef int          v8i  __attribute__((ext_vector_type(8)));

#if defined(__has_builtin)
#if __has_builtin(__builtin_amdgcn_tensor_load_to_lds)
#define USE_TDM 1
#endif
#endif
#ifndef USE_TDM
#define USE_TDM 0
#endif

// ---------- bf16 <-> f32 helpers (round to nearest even) ----------
__device__ __forceinline__ bf16 f2bf(float f) {
    union { float f; uint32_t u; } v; v.f = f;
    uint32_t r = v.u + 0x7FFFu + ((v.u >> 16) & 1u);
    uint16_t h = (uint16_t)(r >> 16);
    return __builtin_bit_cast(bf16, h);
}

// ---------- WMMA fragment loaders (wave32, 16x16x32 bf16) ----------
// A-matrix 16x32 (MxK): lane L -> row M=L&15, elems 0..7: K=half*8+i,
// elems 8..15: K=16+half*8+(i-8).  Two 16B loads per lane.
__device__ __forceinline__ v16bf load_A(const bf16* src, int ld) {
    int lane = threadIdx.x & 31;
    int m = lane & 15, lh = lane >> 4;
    const bf16* row = src + (size_t)m * ld + lh * 8;
    v8bf lo = *(const v8bf*)(row);
    v8bf hi = *(const v8bf*)(row + 16);
    v16bf a;
#pragma unroll
    for (int i = 0; i < 8; ++i) { a[i] = lo[i]; a[i + 8] = hi[i]; }
    return a;
}
// B-matrix 32x16 (KxN) from row-major B^T storage src[n][k] (k contiguous):
// lane: N=lane&15, elems 0..7: K=half*8+i, elems 8..15: K=16+half*8+(i-8).  Two 16B loads per lane.
__device__ __forceinline__ v16bf load_Bt(const bf16* src, int ld) {
    int lane = threadIdx.x & 31;
    int n = lane & 15, lh = lane >> 4;
    const bf16* row = src + (size_t)n * ld + lh * 8;
    v8bf lo = *(const v8bf*)(row);
    v8bf hi = *(const v8bf*)(row + 16);
    v16bf b;
#pragma unroll
    for (int i = 0; i < 8; ++i) { b[i] = lo[i]; b[i + 8] = hi[i]; }
    return b;
}

__device__ __forceinline__ v8f wmma_bf16(v16bf a, v16bf b, v8f c) {
    return __builtin_amdgcn_wmma_f32_16x16x32_bf16(
        /*neg_a=*/false, a, /*neg_b=*/false, b,
        /*c_mod=*/(short)0, c, /*reuse_a=*/false, /*reuse_b=*/false);
}

// ---------- kernel 1: f32 -> bf16 convert ----------
__global__ void f32_to_bf16_k(const float* __restrict__ src,
                              bf16* __restrict__ dst, int n) {
    for (int i = blockIdx.x * blockDim.x + threadIdx.x; i < n;
         i += gridDim.x * blockDim.x)
        dst[i] = f2bf(src[i]);
}

// ---------- kernel 2: NT GEMM  C[MxN] = A[MxK] * B[NxK]^T (bf16 -> f32) ----
// 8 waves; each wave owns a 32x64 tile (2 A-frags x 4 B-frags = 8 WMMAs/k-step)
#define BM 256
#define BN 64
#define BK 32
__global__ __launch_bounds__(256) void gemm_nt_bf16(
    const bf16* __restrict__ A, const bf16* __restrict__ B,
    float* __restrict__ C, int M, int N, int K) {
    __shared__ bf16 As[BM][BK + 8];   // 20 KB
    __shared__ bf16 Bs[BN][BK + 8];   //  5 KB
    int tid = threadIdx.x;
    int wave = tid >> 5;
    int m0 = blockIdx.x * BM;
    int n0 = blockIdx.y * BN;
    v8f acc[2][4] = {};
    for (int k0 = 0; k0 < K; k0 += BK) {
        // stage A tile (256x32): 1024 16B chunks, 4 per thread
#pragma unroll
        for (int it = 0; it < 4; ++it) {
            int idx = tid + it * 256;
            int r = idx >> 2, s = (idx & 3) * 8;
            *(uint4*)(&As[r][s]) = *(const uint4*)(A + (size_t)(m0 + r) * K + k0 + s);
            if (k0 + BK < K)
                __builtin_prefetch(A + (size_t)(m0 + r) * K + k0 + BK + s, 0, 3);
        }
        // stage B tile (64x32): 256 16B chunks, 1 per thread
        {
            int r = tid >> 2, s = (tid & 3) * 8;
            *(uint4*)(&Bs[r][s]) = *(const uint4*)(B + (size_t)(n0 + r) * K + k0 + s);
            if (k0 + BK < K)
                __builtin_prefetch(B + (size_t)(n0 + r) * K + k0 + BK + s, 0, 3);
        }
        __syncthreads();
        v16bf bfr[4];
#pragma unroll
        for (int nt = 0; nt < 4; ++nt)
            bfr[nt] = load_Bt(&Bs[nt * 16][0], BK + 8);
#pragma unroll
        for (int mt = 0; mt < 2; ++mt) {
            v16bf a = load_A(&As[wave * 32 + mt * 16][0], BK + 8);
#pragma unroll
            for (int nt = 0; nt < 4; ++nt)
                acc[mt][nt] = wmma_bf16(a, bfr[nt], acc[mt][nt]);
        }
        __syncthreads();
    }
    // C/D layout: VGPR r, lanes 0-15: M=r, N=lane; lanes 16-31: M=r+8
    int lane = tid & 31, lh = lane >> 4, n = lane & 15;
#pragma unroll
    for (int mt = 0; mt < 2; ++mt)
#pragma unroll
        for (int nt = 0; nt < 4; ++nt)
#pragma unroll
            for (int r = 0; r < 8; ++r) {
                int m = m0 + wave * 32 + mt * 16 + r + 8 * lh;
                C[(size_t)m * N + n0 + nt * 16 + n] = acc[mt][nt][r];
            }
}

// ---------- kernel 3: RMSNorm + RoPE (+gain) / convert, wave per head-row --
// src: f32 [B*T, nH*HD]   dst: bf16 [B, nH, T, HD]
// mode 0: q (norm+rope+gain)  1: k (norm+rope)  2: v (convert only)
__global__ __launch_bounds__(128) void norm_rope_k(
    const float* __restrict__ src, bf16* __restrict__ dst,
    const float* __restrict__ gain, int nH, int T, int mode) {
    int gtid = blockIdx.x * 128 + threadIdx.x;
    int wid = gtid >> 5;
    int lane = threadIdx.x & 31;
    int h = wid % nH;
    int t = (wid / nH) % T;
    int b = wid / (nH * T);
    const float* row = src + (size_t)wid * HD;
    bf16* out = dst + (((size_t)(b * nH + h) * T) + t) * HD;
    if (mode == 2) {
#pragma unroll
        for (int j = 0; j < 4; ++j) {
            int d = lane * 4 + j;
            out[d] = f2bf(row[d]);
        }
        return;
    }
    int pidx[2] = {lane, lane + 32};
    float te[2], to[2];
    float sum = 0.f;
#pragma unroll
    for (int j = 0; j < 2; ++j) {
        te[j] = row[2 * pidx[j]];
        to[j] = row[2 * pidx[j] + 1];
        sum += te[j] * te[j] + to[j] * to[j];
    }
#pragma unroll
    for (int off = 1; off < 32; off <<= 1)
        sum += __shfl_xor(sum, off, 32);
    float rinv = rsqrtf(sum * (1.0f / 128.0f) + 1.1920928955078125e-07f);
    float g = (mode == 0) ? gain[h] : 1.0f;
#pragma unroll
    for (int j = 0; j < 2; ++j) {
        float fr = (float)t * __powf(10000.0f, -(float)(2 * pidx[j]) * (1.0f / 128.0f));
        float c = __cosf(fr), s = __sinf(fr);
        float ten = te[j] * rinv, ton = to[j] * rinv;
        out[2 * pidx[j]]     = f2bf((ten * c - ton * s) * g);
        out[2 * pidx[j] + 1] = f2bf((ten * s + ton * c) * g);
    }
}

// ---------- kernel 4: flash attention (causal, online softmax) -------------
// qh: [B,H,T,HD]  kh/vh: [B,HKV,T,HD]  attn out: [B,T,H*HD]  (all bf16)
__global__ __launch_bounds__(128) void flash_attn_k(
    const bf16* __restrict__ qh, const bf16* __restrict__ kh,
    const bf16* __restrict__ vh, bf16* __restrict__ attn, int T) {
    __shared__ bf16 Kt[32][HD + 8];       // keys row-major (d contiguous), TDM-padded
    __shared__ bf16 Vt[HD][32 + 8];       // V transposed (key contiguous)
    __shared__ bf16 Pw[4][16][32 + 8];    // per-wave P staging
    int tid = threadIdx.x, wave = tid >> 5, lane = tid & 31;
    int lh = lane >> 4, nn = lane & 15;
    int bh = blockIdx.x;
    int b = bh / H, h = bh % H, hkv = h / (H / HKV);
    int q0 = blockIdx.y * 64;
    int qw = q0 + wave * 16;
    const bf16* Qbase = qh + ((size_t)(b * H + h) * T + qw) * HD;
    const bf16* Kbase = kh + ((size_t)(b * HKV + hkv) * T) * HD;
    const bf16* Vbase = vh + ((size_t)(b * HKV + hkv) * T) * HD;

    v16bf qa[4];
#pragma unroll
    for (int kk = 0; kk < 4; ++kk) qa[kk] = load_A(Qbase + kk * 32, HD);

    v8f acc[8] = {};
    float m_i[8], l_i[8];
#pragma unroll
    for (int r = 0; r < 8; ++r) { m_i[r] = -1e30f; l_i[r] = 0.f; }
    const float scale = 0.08838834764831845f;   // 1/sqrt(128)

    int kb_end = (q0 + 64) / 32;
    for (int kb = 0; kb < kb_end; ++kb) {
        int kbase = kb * 32;
#if USE_TDM
        // --- K tile via Tensor Data Mover: 2D tile 128x32 (2B elems),
        // LDS padding 4 DWORDs every 64 DWORDs reproduces the HD+8 stride.
        if (wave == 0) {
            uint64_t ga = (uint64_t)(uintptr_t)(Kbase + (size_t)kbase * HD);
            uint32_t la = (uint32_t)(uintptr_t)(&Kt[0][0]);
            v4u g0;
            g0[0] = 1u;                                   // count=1 (valid D#)
            g0[1] = la;                                   // lds_addr
            g0[2] = (uint32_t)ga;                         // global_addr[31:0]
            g0[3] = (uint32_t)((ga >> 32) & 0x01FFFFFFu)  // global_addr[56:32]
                    | (2u << 30);                         // type=2 ("image")
            v8i g1;
            g1[0] = (int)((1u << 16)    // data_size=1 -> 2 bytes
                  | (1u << 20)          // pad_enable
                  | (5u << 22)          // pad_interval: 64 DWORDs
                  | (3u << 25));        // pad_amount: 4 DWORDs (16B)
            g1[1] = (int)((uint32_t)HD << 16);   // tensor_dim0 low bits
            g1[2] = (int)((uint32_t)T  << 16);   // tensor_dim1 low bits
            g1[3] = (int)((uint32_t)HD << 16);   // tile_dim0 = 128 at [127:112]
            g1[4] = 32;                          // tile_dim1 = 32 rows
            g1[5] = HD;                          // tensor_dim0_stride = HD
            g1[6] = 0;
            g1[7] = 0;
            v4i gz4 = {0, 0, 0, 0};              // 2D tensor: groups 2/3 unused
            v8i gz8 = {0, 0, 0, 0, 0, 0, 0, 0};  // extra group (clang-23 6-arg form)
            __builtin_amdgcn_tensor_load_to_lds(g0, g1, gz4, gz4, gz8, 0);
        }
#else
        // stage K tile 32x128 (16B chunks)
#pragma unroll
        for (int it = 0; it < 4; ++it) {
            int idx = tid + it * 128;
            int r = idx >> 4, c = (idx & 15) * 8;
            *(uint4*)(&Kt[r][c]) =
                *(const uint4*)(Kbase + (size_t)(kbase + r) * HD + c);
        }
#endif
        // stage V tile transposed (manual: TDM cannot transpose)
#pragma unroll
        for (int it = 0; it < 4; ++it) {
            int idx = tid + it * 128;
            int r = idx >> 4, c = (idx & 15) * 8;
            uint4 raw = *(const uint4*)(Vbase + (size_t)(kbase + r) * HD + c);
            bf16 tv[8];
            *(uint4*)tv = raw;
#pragma unroll
            for (int j = 0; j < 8; ++j) Vt[c + j][r] = tv[j];
        }
#if USE_TDM
        if (wave == 0) __builtin_amdgcn_s_wait_tensorcnt(0);
#endif
        __syncthreads();

        // S = Q * K^T  (two 16-key N-tiles, K-dim = HD via 4 steps)
        v8f s[2] = {};
#pragma unroll
        for (int nt = 0; nt < 2; ++nt)
#pragma unroll
            for (int kk = 0; kk < 4; ++kk) {
                v16bf bfr = load_Bt(&Kt[nt * 16][kk * 32], HD + 8);
                s[nt] = wmma_bf16(qa[kk], bfr, s[nt]);
            }

        // causal mask + online softmax (row stats via 16-lane shuffles)
#pragma unroll
        for (int r = 0; r < 8; ++r) {
            int qidx = qw + r + 8 * lh;
#pragma unroll
            for (int nt = 0; nt < 2; ++nt) {
                int kidx = kbase + nt * 16 + nn;
                float val = s[nt][r] * scale;
                if (kidx > qidx) val = -1e30f;
                s[nt][r] = val;
            }
            float mx = fmaxf(s[0][r], s[1][r]);
#pragma unroll
            for (int off = 1; off < 16; off <<= 1)
                mx = fmaxf(mx, __shfl_xor(mx, off, 32));
            float m_new = fmaxf(m_i[r], mx);
            float alpha = __expf(m_i[r] - m_new);
            float p0 = __expf(s[0][r] - m_new);
            float p1 = __expf(s[1][r] - m_new);
            float ps = p0 + p1;
#pragma unroll
            for (int off = 1; off < 16; off <<= 1)
                ps += __shfl_xor(ps, off, 32);
            l_i[r] = l_i[r] * alpha + ps;
            m_i[r] = m_new;
#pragma unroll
            for (int d = 0; d < 8; ++d) acc[d][r] *= alpha;
            // C-layout -> LDS for A-layout reload
            Pw[wave][r + 8 * lh][nn]      = f2bf(p0);
            Pw[wave][r + 8 * lh][16 + nn] = f2bf(p1);
        }

        // O += P * V  (P 16x32, V 32x16 per HD tile)
        v16bf pa = load_A(&Pw[wave][0][0], 32 + 8);
#pragma unroll
        for (int nt = 0; nt < 8; ++nt) {
            v16bf bfr = load_Bt(&Vt[nt * 16][0], 32 + 8);
            acc[nt] = wmma_bf16(pa, bfr, acc[nt]);
        }
        __syncthreads();
    }

    // epilogue: normalize, write bf16 [B,T,H*HD]
#pragma unroll
    for (int nt = 0; nt < 8; ++nt)
#pragma unroll
        for (int r = 0; r < 8; ++r) {
            int q = qw + r + 8 * lh;
            float o = acc[nt][r] / l_i[r];
            attn[((size_t)(b * T + q)) * (H * HD) + h * HD + nt * 16 + nn] = f2bf(o);
        }
}

// ---------- launch ---------------------------------------------------------
extern "C" void kernel_launch(void* const* d_in, const int* in_sizes, int n_in,
                              void* d_out, int out_size, void* d_ws, size_t ws_size,
                              hipStream_t stream) {
    (void)in_sizes; (void)n_in; (void)out_size; (void)ws_size;
    const float* x    = (const float*)d_in[0];
    const float* Wq   = (const float*)d_in[1];
    const float* Wk   = (const float*)d_in[2];
    const float* Wv   = (const float*)d_in[3];
    const float* Wo   = (const float*)d_in[4];
    const float* gain = (const float*)d_in[5];
    float* out = (float*)d_out;

    constexpr int B = 2, T = 2048, C = 2048;
    constexpr int M  = B * T;            // 4096
    constexpr int NK = HKV * HD;         // 512
    const size_t nx  = (size_t)M * C;    // 8388608
    const size_t nwq = (size_t)C * C;
    const size_t nwk = (size_t)NK * C;

    // workspace carve-up (256B aligned)
    size_t off = 0;
    auto carve = [&](size_t bytes) {
        void* p = (char*)d_ws + off;
        off += (bytes + 255) & ~(size_t)255;
        return p;
    };
    bf16*  xb   = (bf16*)carve(nx * 2);
    bf16*  Wqb  = (bf16*)carve(nwq * 2);
    bf16*  Wkb  = (bf16*)carve(nwk * 2);
    bf16*  Wvb  = (bf16*)carve(nwk * 2);
    bf16*  Wob  = (bf16*)carve(nwq * 2);
    float* qf   = (float*)carve(nx * 4);
    float* kf   = (float*)carve((size_t)M * NK * 4);
    float* vf   = (float*)carve((size_t)M * NK * 4);
    bf16*  qhb  = (bf16*)carve(nx * 2);
    bf16*  khb  = (bf16*)carve((size_t)M * NK * 2);
    bf16*  vhb  = (bf16*)carve((size_t)M * NK * 2);
    bf16*  attn = (bf16*)carve(nx * 2);

    // 1) convert inputs to bf16
    f32_to_bf16_k<<<4096, 256, 0, stream>>>(x,  xb,  (int)nx);
    f32_to_bf16_k<<<4096, 256, 0, stream>>>(Wq, Wqb, (int)nwq);
    f32_to_bf16_k<<<2048, 256, 0, stream>>>(Wk, Wkb, (int)nwk);
    f32_to_bf16_k<<<2048, 256, 0, stream>>>(Wv, Wvb, (int)nwk);
    f32_to_bf16_k<<<4096, 256, 0, stream>>>(Wo, Wob, (int)nwq);

    // 2) projections: q = x Wq^T, k = x Wk^T, v = x Wv^T
    gemm_nt_bf16<<<dim3(M / BM, C  / BN), 256, 0, stream>>>(xb, Wqb, qf, M, C,  C);
    gemm_nt_bf16<<<dim3(M / BM, NK / BN), 256, 0, stream>>>(xb, Wkb, kf, M, NK, C);
    gemm_nt_bf16<<<dim3(M / BM, NK / BN), 256, 0, stream>>>(xb, Wvb, vf, M, NK, C);

    // 3) RMSNorm + RoPE (+gain) / convert, relayout to [B,h,T,HD]
    norm_rope_k<<<(M * H)   / 4, 128, 0, stream>>>(qf, qhb, gain, H,   T, 0);
    norm_rope_k<<<(M * HKV) / 4, 128, 0, stream>>>(kf, khb, gain, HKV, T, 1);
    norm_rope_k<<<(M * HKV) / 4, 128, 0, stream>>>(vf, vhb, gain, HKV, T, 2);

    // 4) causal flash attention
    flash_attn_k<<<dim3(B * H, T / 64), 128, 0, stream>>>(qhb, khb, vhb, attn, T);

    // 5) output projection: out = attn Wo^T  (f32 result)
    gemm_nt_bf16<<<dim3(M / BM, C / BN), 256, 0, stream>>>(attn, Wob, out, M, C, C);
}
